// VQ_layer_18769007084529
// MI455X (gfx1250) — matrix operand order repaired and checked by reading, compiled, not run on hardware
//
#include <hip/hip_runtime.h>
#include <hip/hip_bf16.h>

typedef __attribute__((ext_vector_type(16))) _Float16 v16h;
typedef __attribute__((ext_vector_type(8)))  _Float16 v8h;
typedef __attribute__((ext_vector_type(8)))  float    v8f;
typedef __attribute__((ext_vector_type(4)))  unsigned int u32x4;
typedef __attribute__((ext_vector_type(8)))  int          i32x8;
typedef __attribute__((ext_vector_type(4)))  int          i32x4;

#define DIM     64      // embedding dim (reduction axis)
#define KCODES  512     // codebook size
#define COLPAD  72      // halves per LDS column (64 + 8 pad -> 144B stride, bank-friendly)
#define BETA    0.25f

// ---------------------------------------------------------------------------
// Kernel 0: convert codebook to f16 column-major in ws, per-code ||e_k||^2,
// zero the loss slot. Runs once; main kernel blocks then DMA the f16 table.
// ---------------------------------------------------------------------------
__global__ void vq_prep(const float* __restrict__ emb,
                        _Float16* __restrict__ ebf16,    // ws: [k][d], k-major
                        float* __restrict__ enorm,
                        float* __restrict__ loss_slot) {
    int k = blockIdx.x * 256 + threadIdx.x;
    if (k < KCODES) {
        float s = 0.0f;
        #pragma unroll 8
        for (int d = 0; d < DIM; ++d) {
            float e = emb[d * KCODES + k];       // coalesced across k
            s += e * e;
            ebf16[k * DIM + d] = (_Float16)e;    // column-major (contiguous per code)
        }
        enorm[k] = s;
    }
    if (k == 0) *loss_slot = 0.0f;
}

// ---------------------------------------------------------------------------
// Kernel 1: distances via WMMA f16->f32, argmin, gather, store, loss.
// Grid: n_rows/128 blocks, 256 threads (8 wave32's), 16 rows per wave.
// Codebook staged into LDS by the Tensor Data Mover with HW padding.
// ---------------------------------------------------------------------------
__global__ __launch_bounds__(256) void vq_main(const float* __restrict__ x,
                                               const float* __restrict__ emb,
                                               const _Float16* __restrict__ ebf16_g,
                                               const float* __restrict__ enorm_g,
                                               float* __restrict__ out_q,
                                               float* __restrict__ loss_out,
                                               float loss_scale) {
    // Codebook as f16, column-major with pad: ebcol[k*COLPAD + d]
    __shared__ __align__(16) _Float16 ebcol[KCODES * COLPAD];
    __shared__ float enorm_s[KCODES];

    const int tid = threadIdx.x;

#if __has_builtin(__builtin_amdgcn_tensor_load_to_lds)
    // --- TDM staging: one descriptor, wave 0 only. The pad feature inserts
    // 16B after every 128B stored, producing the COLPAD=72 layout in HW. ---
    if (tid < 32) {
        typedef __attribute__((address_space(3))) _Float16 lds_h;
        unsigned lds_off = (unsigned)(uintptr_t)(lds_h*)&ebcol[0];
        unsigned long long ga = (unsigned long long)(uintptr_t)ebf16_g;

        u32x4 g0;
        g0[0] = 1u;                                   // count=1 (valid descriptor)
        g0[1] = lds_off;                              // lds_addr (bytes)
        g0[2] = (unsigned)(ga & 0xFFFFFFFFu);         // global_addr[31:0]
        g0[3] = (unsigned)((ga >> 32) & 0x1FFFFFFu)   // global_addr[56:32]
              | (2u << 30);                           // type=2 ("image")

        i32x8 g1;
        g1[0] = (int)((1u << 16)    // data_size: 2 bytes
                    | (1u << 20)    // pad_enable
                    | (4u << 22)    // pad_interval: 8*2^4 = 128 bytes
                    | (3u << 25));  // pad_amount: 4 DWORDs = 16 bytes
        g1[1] = (int)(((unsigned)(DIM * KCODES) & 0xFFFFu) << 16); // tensor_dim0 lo16
        g1[2] = (int)((((unsigned)(DIM * KCODES)) >> 16)           // tensor_dim0 hi16
                    | (1u << 16));                                 // tensor_dim1 = 1
        g1[3] = (int)(((unsigned)(DIM * KCODES) & 0xFFFFu) << 16); // tile_dim0 = 32768
        g1[4] = 1;                                                 // tile_dim1 = 1
        g1[5] = DIM * KCODES;                                      // tensor_dim0_stride
        g1[6] = 0;
        g1[7] = 0;

        i32x4 gz4 = {0, 0, 0, 0};                     // <=2D: groups 2/3 unused
        i32x8 gz8 = {0, 0, 0, 0, 0, 0, 0, 0};         // trailing block unused
        __builtin_amdgcn_tensor_load_to_lds(g0, g1, gz4, gz4, gz8, 0);
        __builtin_amdgcn_s_wait_tensorcnt(0);
    }
#else
    // Fallback: manual staging from the pre-converted f16 table (coalesced).
    for (int i = tid; i < DIM * KCODES; i += 256) {
        int k = i >> 6, d = i & (DIM - 1);
        ebcol[k * COLPAD + d] = ebf16_g[i];
    }
#endif
    for (int i = tid; i < KCODES; i += 256) enorm_s[i] = enorm_g[i];
    __syncthreads();

    const int wave = tid >> 5;        // 0..7
    const int lane = tid & 31;
    const int l16  = lane & 15;
    const int hgrp = lane >> 4;       // 0 = lanes 0-15, 1 = lanes 16-31
    const int rowbase = blockIdx.x * 128 + wave * 16;

    // ---- A fragments: 16 rows x 64 d, as two 16x32 f16 tiles --------------
    // ISA 16-bit A 16x32 layout: lanes 0-15 hold K=0..7 (v0-3) and 16..23 (v4-7);
    // lanes 16-31 hold K=8..15 and 24..31 (same M rows).
    v16h a0, a1;
    {
        const float4* xr4 =
            reinterpret_cast<const float4*>(x + (size_t)(rowbase + l16) * DIM);
        float4 f[8];
        f[0] = xr4[2 * hgrp + 0];
        f[1] = xr4[2 * hgrp + 1];
        f[2] = xr4[4 + 2 * hgrp + 0];
        f[3] = xr4[4 + 2 * hgrp + 1];
        f[4] = xr4[8 + 2 * hgrp + 0];
        f[5] = xr4[8 + 2 * hgrp + 1];
        f[6] = xr4[12 + 2 * hgrp + 0];
        f[7] = xr4[12 + 2 * hgrp + 1];
        const float* fp = reinterpret_cast<const float*>(f);
        #pragma unroll
        for (int j = 0; j < 16; ++j) {
            a0[j] = (_Float16)fp[j];
            a1[j] = (_Float16)fp[16 + j];
        }
    }

    // ---- sweep code tiles 2-at-a-time; second tile's WMMAs hide the
    // first tile's WMMA->VALU hazard window --------------------------------
    float minv[8];
    int   mini[8];
    #pragma unroll
    for (int v = 0; v < 8; ++v) { minv[v] = 3.4e38f; mini[v] = 0; }

    for (int t = 0; t < KCODES / 16; t += 2) {
        const int colA = t * 16 + l16;
        const int colB = colA + 16;
        const _Float16* cbA = &ebcol[colA * COLPAD];
        const _Float16* cbB = &ebcol[colB * COLPAD];

        // B 32x16 layout: lanes 0-15 hold K(=d)=0..7,16..23; lanes 16-31 hold 8..15,24..31.
        v8h qa0 = *(const v8h*)(cbA + 8 * hgrp);
        v8h qa1 = *(const v8h*)(cbA + 16 + 8 * hgrp);
        v8h qa2 = *(const v8h*)(cbA + 32 + 8 * hgrp);
        v8h qa3 = *(const v8h*)(cbA + 48 + 8 * hgrp);
        v8h qb0 = *(const v8h*)(cbB + 8 * hgrp);
        v8h qb1 = *(const v8h*)(cbB + 16 + 8 * hgrp);
        v8h qb2 = *(const v8h*)(cbB + 32 + 8 * hgrp);
        v8h qb3 = *(const v8h*)(cbB + 48 + 8 * hgrp);
        v16h b0A, b1A, b0B, b1B;
        #pragma unroll
        for (int j = 0; j < 8; ++j) {
            b0A[j] = qa0[j]; b0A[8 + j] = qa1[j];
            b1A[j] = qa2[j]; b1A[8 + j] = qa3[j];
            b0B[j] = qb0[j]; b0B[8 + j] = qb1[j];
            b1B[j] = qb2[j]; b1B[8 + j] = qb3[j];
        }

        v8f cA = {};
        cA = __builtin_amdgcn_wmma_f32_16x16x32_f16(false, a0, false, b0A,
                                                    (short)0, cA, false, false);
        cA = __builtin_amdgcn_wmma_f32_16x16x32_f16(false, a1, false, b1A,
                                                    (short)0, cA, false, false);
        v8f cB = {};
        cB = __builtin_amdgcn_wmma_f32_16x16x32_f16(false, a0, false, b0B,
                                                    (short)0, cB, false, false);
        cB = __builtin_amdgcn_wmma_f32_16x16x32_f16(false, a1, false, b1B,
                                                    (short)0, cB, false, false);

        const float enA = enorm_s[colA];
        const float enB = enorm_s[colB];
        #pragma unroll
        for (int v = 0; v < 8; ++v) {
            float dA = enA - 2.0f * cA[v];
            if (dA < minv[v]) { minv[v] = dA; mini[v] = colA; }
        }
        #pragma unroll
        for (int v = 0; v < 8; ++v) {
            float dB = enB - 2.0f * cB[v];
            if (dB < minv[v]) { minv[v] = dB; mini[v] = colB; }
        }
    }

    // ---- per-row argmin: butterfly across the 16 lanes of each half -------
    #pragma unroll
    for (int v = 0; v < 8; ++v) {
        float mv = minv[v];
        int   mi = mini[v];
        #pragma unroll
        for (int m = 1; m < 16; m <<= 1) {   // xor masks < 16 stay within half
            float ov = __shfl_xor(mv, m, 32);
            int   oi = __shfl_xor(mi, m, 32);
            if (ov < mv || (ov == mv && oi < mi)) { mv = ov; mi = oi; }
        }
        mini[v] = mi;   // uniform across the 16 lanes of this half
    }

    // ---- gather winning code (exact f32), store quantized_st, loss --------
    float lacc = 0.0f;
    #pragma unroll
    for (int v = 0; v < 8; ++v) {
        const int row = rowbase + v + 8 * hgrp;
        const int k   = mini[v];
        const int d0  = l16 * 4;            // this lane handles d0..d0+3
        float q0 = emb[(d0 + 0) * KCODES + k];
        float q1 = emb[(d0 + 1) * KCODES + k];
        float q2 = emb[(d0 + 2) * KCODES + k];
        float q3 = emb[(d0 + 3) * KCODES + k];
        const float4 xv =
            reinterpret_cast<const float4*>(x + (size_t)row * DIM)[l16];
        float e0 = q0 - xv.x, e1 = q1 - xv.y, e2 = q2 - xv.z, e3 = q3 - xv.w;
        lacc += e0 * e0 + e1 * e1 + e2 * e2 + e3 * e3;
        float4 qv = make_float4(q0, q1, q2, q3);
        reinterpret_cast<float4*>(out_q + (size_t)row * DIM)[l16] = qv;
    }

    #pragma unroll
    for (int m = 1; m < 32; m <<= 1) lacc += __shfl_xor(lacc, m, 32);
    if (lane == 0) atomicAdd(loss_out, lacc * loss_scale);
}

// ---------------------------------------------------------------------------
extern "C" void kernel_launch(void* const* d_in, const int* in_sizes, int n_in,
                              void* d_out, int out_size, void* d_ws, size_t ws_size,
                              hipStream_t stream) {
    const float* x   = (const float*)d_in[0];   // [65536, 64] flattened
    const float* emb = (const float*)d_in[1];   // [64, 512]
    float* out       = (float*)d_out;           // quantized_st [N*64] then loss [1]
    float* loss_ptr  = out + (out_size - 1);

    // ws layout: [0,64KB) f16 codebook col-major; [64KB,66KB) enorm f32[512]
    _Float16* ebf16_ws = (_Float16*)d_ws;
    float*    enorm_ws = (float*)((char*)d_ws + (size_t)DIM * KCODES * sizeof(_Float16));

    const int n_rows = in_sizes[0] / DIM;               // 65536
    const float loss_scale = (1.0f + BETA) / (float)in_sizes[0];

    vq_prep<<<(KCODES + 255) / 256, 256, 0, stream>>>(emb, ebf16_ws, enorm_ws,
                                                      loss_ptr);
    vq_main<<<n_rows / 128, 256, 0, stream>>>(x, emb, ebf16_ws, enorm_ws, out,
                                              loss_ptr, loss_scale);
}
